// BinaryTreeCell_65017214926868
// MI455X (gfx1250) — matrix operand order repaired and checked by reading, not compiled, not run on hardware
//
#include <hip/hip_runtime.h>
#include <hip/hip_bf16.h>

typedef __attribute__((ext_vector_type(16))) __bf16 v16bf;
typedef __attribute__((ext_vector_type(8)))  float  v8f;
typedef int v4i_vs __attribute__((vector_size(16)));   // builtin's b128 payload type

#define D_DIM 256
#define WAVES 8
#define ROWS_PER_WAVE 16
#define ROWS_PER_BLOCK (WAVES * ROWS_PER_WAVE)   // 128
#define A_ROW_BYTES 528                          // 256 bf16 (512B) + 16B pad (bank spread)
#define A_MAT_BYTES (16 * A_ROW_BYTES)           // 8448
#define A_TOTAL (WAVES * 3 * A_MAT_BYTES)        // 202752
#define B_TILE_BYTES 1024                        // 16x32 bf16 tile
#define B_TOTAL (14 * 8 * B_TILE_BYTES)          // 114688
#define LDS_TOTAL (A_TOTAL + B_TOTAL)            // 317440  (<= 320KB WGP LDS)

#if defined(__AMDGCN__) && __has_builtin(__builtin_amdgcn_global_load_async_to_lds_b128)
#define USE_ASYNC_B 1
#else
#define USE_ASYNC_B 0
#endif

// ---------- helpers ----------
__device__ __forceinline__ unsigned int pk_bf16(float a, float b) {
    unsigned ua = __float_as_uint(a);
    unsigned ub = __float_as_uint(b);
    ua = (ua + 0x7FFFu + ((ua >> 16) & 1u)) >> 16;   // round-to-nearest-even
    ub = (ub + 0x7FFFu + ((ub >> 16) & 1u)) >> 16;
    return (ub << 16) | (ua & 0xFFFFu);
}

__device__ __forceinline__ float fast_exp(float x) {
    return __builtin_amdgcn_exp2f(x * 1.44269504f);
}
__device__ __forceinline__ float fsig(float x) {
    return __builtin_amdgcn_rcpf(1.0f + fast_exp(-x));
}
__device__ __forceinline__ float ftanh(float x) {
    return 2.0f * fsig(2.0f * x) - 1.0f;
}

__device__ __forceinline__ void wait_async_zero() {
#if USE_ASYNC_B
#if __has_builtin(__builtin_amdgcn_s_wait_asynccnt)
    __builtin_amdgcn_s_wait_asynccnt(0);
#else
    asm volatile("s_wait_asynccnt 0x0" ::: "memory");
#endif
#endif
}

// A-tile (16x32 bf16, MxK) from LDS row-major bf16 rows.
// lane: m = lane&15, kh = lane>>4. VGPR0-3: K = kh*16+0..7 ; VGPR4-7: K = kh*16+8..15
__device__ __forceinline__ v16bf load_a(const char* mBase, int ks, int m, int kh) {
    union { uint4 q[2]; v16bf v; } u;
    const char* rowp = mBase + m * A_ROW_BYTES;
    int k0 = ks * 32 + kh * 16;
    u.q[0] = *(const uint4*)(rowp + (size_t)k0 * 2);
    u.q[1] = *(const uint4*)(rowp + (size_t)(k0 + 8) * 2);
    return u.v;
}

// B-tile (32x16 bf16, KxN) from LDS stage buffer: lane l holds its 32B chunk at l*32.
__device__ __forceinline__ v16bf load_b(const char* bBase, int g, int ks, int lane) {
    union { uint4 q[2]; v16bf v; } u;
    const char* p = bBase + (g * 8 + ks) * B_TILE_BYTES + lane * 32;
    u.q[0] = *(const uint4*)p;
    u.q[1] = *(const uint4*)(p + 16);
    return u.v;
}

// ---------- weight fp32 -> bf16 pre-pass ----------
__global__ void cvt_w_kernel(const float* __restrict__ src,
                             unsigned short* __restrict__ dst, int n) {
    int i = blockIdx.x * blockDim.x + threadIdx.x;
    if (i < n) {
        unsigned u = __float_as_uint(src[i]);
        dst[i] = (unsigned short)((u + 0x7FFFu + ((u >> 16) & 1u)) >> 16);
    }
}

// ---------- fused TreeLSTM cell ----------
// Wb blob (bf16, row-major [14][256][256]), gate order:
//  0:cx 1:ix 2:fx 3:ox | 4:ul 5:il 6:lfl 7:rfl 8:ol | 9:ur 10:ir 11:lfr 12:rfr 13:or
__global__ __launch_bounds__(256)
void tree_lstm_kernel(const float* __restrict__ x,  const float* __restrict__ lh,
                      const float* __restrict__ rh, const float* __restrict__ lc,
                      const float* __restrict__ rc,
                      const unsigned short* __restrict__ Wb,
                      const float* __restrict__ b_cx, const float* __restrict__ b_ix,
                      const float* __restrict__ b_fx, const float* __restrict__ b_ox,
                      float* __restrict__ outc, float* __restrict__ outh) {
    extern __shared__ __align__(16) char smem[];
    const int tid  = threadIdx.x;
    const int wave = tid >> 5;
    const int lane = tid & 31;
    const int n_   = lane & 15;     // tile column / row-in-tile
    const int kh   = lane >> 4;     // K half-select
    const int row0 = blockIdx.x * ROWS_PER_BLOCK + wave * ROWS_PER_WAVE;

    char* aBase = smem + (size_t)wave * 3 * A_MAT_BYTES;
    char* bBase = smem + A_TOTAL;

    // Cooperative stage of the 112 B-tiles of one channel block (14 per wave).
    // Per tile: lane l moves the 32B chunk {W row col0+(l&15), K = ks*32+(l>>4)*16 ..+16}.
    auto stageB = [&](int cb) {
        const int wrow = cb * 16 + n_;
#pragma unroll
        for (int t = 0; t < 14; ++t) {
            int id = wave * 14 + t;           // 0..111 covers 14 gates x 8 ksteps
            int g  = id >> 3;
            int ks = id & 7;
            int koff = ks * 32 + kh * 16;
            const char* srcp = (const char*)Wb +
                (((size_t)g * 256 + wrow) * 256 + koff) * 2;
            char* dstp = bBase + (g * 8 + ks) * B_TILE_BYTES + lane * 32;
#if USE_ASYNC_B
            // ASYNCcnt-tracked DMA of 16B per lane from global to LDS
            __builtin_amdgcn_global_load_async_to_lds_b128(
                (v4i_vs __attribute__((address_space(1)))*)srcp,
                (v4i_vs __attribute__((address_space(3)))*)dstp, 0, 0);
            __builtin_amdgcn_global_load_async_to_lds_b128(
                (v4i_vs __attribute__((address_space(1)))*)srcp,
                (v4i_vs __attribute__((address_space(3)))*)dstp, 16, 0);
#else
            const uint4* s4 = (const uint4*)srcp;
            uint4 lo = s4[0], hi = s4[1];
            uint4* d4 = (uint4*)dstp;
            d4[0] = lo; d4[1] = hi;
#endif
        }
    };

    // ---- Phase 1: stream this wave's 16 rows of x/lh/rh into LDS as bf16 ----
#pragma unroll
    for (int mat = 0; mat < 3; ++mat) {
        const float* s = (mat == 0) ? x : ((mat == 1) ? lh : rh);
        s += (size_t)row0 * D_DIM;
        char* mBase = aBase + mat * A_MAT_BYTES;
        for (int r = 0; r < 16; ++r) {
            const float4* g4 = (const float4*)(s + (size_t)r * D_DIM + lane * 8);
            float4 f0 = g4[0], f1 = g4[1];
            uint4 p;
            p.x = pk_bf16(f0.x, f0.y); p.y = pk_bf16(f0.z, f0.w);
            p.z = pk_bf16(f1.x, f1.y); p.w = pk_bf16(f1.z, f1.w);
            *(uint4*)(mBase + r * A_ROW_BYTES + lane * 16) = p;
        }
    }

    // Prime the B pipeline with channel block 0.
    stageB(0);
    wait_async_zero();
    __syncthreads();

    // ---- Phase 2: per 16-channel block, 14 fused GEMM tiles + gate math ----
    for (int cb = 0; cb < 16; ++cb) {
        const int col0 = cb * 16;

        v8f acc[14];
#pragma unroll
        for (int g = 0; g < 14; ++g) acc[g] = (v8f){0.f,0.f,0.f,0.f,0.f,0.f,0.f,0.f};

#pragma unroll
        for (int ks = 0; ks < 8; ++ks) {
            v16bf ax = load_a(aBase + 0 * A_MAT_BYTES, ks, n_, kh);
            v16bf al = load_a(aBase + 1 * A_MAT_BYTES, ks, n_, kh);
            v16bf ar = load_a(aBase + 2 * A_MAT_BYTES, ks, n_, kh);
#pragma unroll
            for (int g = 0; g < 14; ++g) {
                v16bf b = load_b(bBase, g, ks, lane);
                v16bf a;
                if (g < 4) a = ax; else if (g < 9) a = al; else a = ar;
                acc[g] = __builtin_amdgcn_wmma_f32_16x16x32_bf16(
                    false, a, false, b, (short)0, acc[g], false, false);
            }
        }

        // All waves finished reading this channel block's B tiles.
        __syncthreads();

        // Kick off the next channel block's weight DMA; it overlaps the
        // HBM-bound elementwise combine below.
        if (cb < 15) stageB(cb + 1);

        // ---- elementwise combine; C/D layout: VGPR r -> row = r + 8*kh, col = n_ ----
        {
            const float bcx = b_cx[col0 + n_];
            const float bix = b_ix[col0 + n_];
            const float bfx = b_fx[col0 + n_];
            const float box = b_ox[col0 + n_];
#pragma unroll
            for (int r = 0; r < 8; ++r) {
                const int row = row0 + r + 8 * kh;
                const size_t idx = (size_t)row * D_DIM + col0 + n_;
                const float lcv = lc[idx];
                const float rcv = rc[idx];
                const float u  = ftanh(acc[0][r] + bcx + acc[4][r] + acc[9][r]);
                const float i  = fsig (acc[1][r] + bix + acc[5][r] + acc[10][r]);
                const float lf = fsig (acc[2][r] + bfx + acc[6][r] + acc[11][r]);
                const float rf = fsig (acc[2][r] + bfx + acc[7][r] + acc[12][r]);
                const float c  = i * u + lf * lcv + rf * rcv;
                const float o  = fsig (acc[3][r] + box + acc[8][r] + acc[13][r]);
                outc[idx] = c;
                outh[idx] = o * ftanh(c);
            }
        }

        wait_async_zero();
        __syncthreads();   // B stage buffer ready for the next channel block
    }
}

extern "C" void kernel_launch(void* const* d_in, const int* in_sizes, int n_in,
                              void* d_out, int out_size, void* d_ws, size_t ws_size,
                              hipStream_t stream) {
    const float* x  = (const float*)d_in[0];
    const float* lc = (const float*)d_in[1];
    const float* lh = (const float*)d_in[2];
    const float* rc = (const float*)d_in[3];
    const float* rh = (const float*)d_in[4];

    // d_in weight indices for gate order {cx,ix,fx,ox, ul,il,lfl,rfl,ol, ur,ir,lfr,rfr,or}
    const int widx[14] = {5, 11, 9, 7, 19, 13, 15, 17, 21, 20, 14, 16, 18, 22};
    unsigned short* Wb = (unsigned short*)d_ws;   // bf16 blob [14][256][256]
    const int WN = D_DIM * D_DIM;                 // 65536 per matrix
    for (int g = 0; g < 14; ++g) {
        cvt_w_kernel<<<WN / 256, 256, 0, stream>>>(
            (const float*)d_in[widx[g]], Wb + (size_t)g * WN, WN);
    }

    const int Nrows = in_sizes[0] / D_DIM;        // 131072
    float* outc = (float*)d_out;
    float* outh = outc + (size_t)Nrows * D_DIM;

    tree_lstm_kernel<<<dim3(Nrows / ROWS_PER_BLOCK), 256, LDS_TOTAL, stream>>>(
        x, lh, rh, lc, rc, Wb,
        (const float*)d_in[6],  /* b_cx */
        (const float*)d_in[12], /* b_ix */
        (const float*)d_in[10], /* b_fx */
        (const float*)d_in[8],  /* b_ox */
        outc, outh);
}
